// CSAEncoder_9431748182065
// MI455X (gfx1250) — matrix operand order repaired and checked by reading, not compiled, not run on hardware
//
#include <hip/hip_runtime.h>

typedef __attribute__((ext_vector_type(16))) _Float16 v16h;
typedef __attribute__((ext_vector_type(8)))  _Float16 v8h;
typedef __attribute__((ext_vector_type(8)))  float    v8f;

#define NB   2      // batch
#define NC   512    // channels
#define NHD  16     // heads
#define ND   32     // head dim
#define NS   1024   // seq (32x32)
#define NW   32     // image width
#define EPSV 1e-5f

// ---------------------------------------------------------------------------
// WMMA fragment helpers (wave32; contiguous k order k = 16*(lane/16) + s in both A and B)
// All loads are contiguous 16B/32B vectors -> global_load_b128 / ds_load_b128.
// ---------------------------------------------------------------------------
__device__ __forceinline__ v8f wmma32(v16h a, v16h b, v8f c) {
  return __builtin_amdgcn_wmma_f32_16x16x32_f16(false, a, false, b, (short)0, c,
                                                false, false);
}

__device__ __forceinline__ v16h ld16(const _Float16* p) {
  v8h lo = *(const v8h*)p;
  v8h hi = *(const v8h*)(p + 16);
  return __builtin_shufflevector(lo, hi, 0, 1, 2, 3, 4, 5, 6, 7,
                                 8, 9, 10, 11, 12, 13, 14, 15);
}

// A fragment: 16x32 f16, row-major source (row0,k0 at A).
// lane<16: row=lane, K 0..15 ; lane>=16: row=lane-16, K 16..31
__device__ __forceinline__ v16h load_a_frag(const _Float16* A, int lda) {
  int lane = threadIdx.x & 31;
  return *(const v16h*)(A + (size_t)(lane & 15) * lda + ((lane < 16) ? 0 : 16));
}

// B fragment from an N-major source: element (k,n) at Bt[n*ldt + k].
// lane<16: col=lane, K 0..15 ; lane>=16: col=lane-16, K 16..31 (contiguous!)
__device__ __forceinline__ v16h load_bT_frag(const _Float16* Bt, int ldt) {
  int lane = threadIdx.x & 31;
  return *(const v16h*)(Bt + (size_t)(lane & 15) * ldt + ((lane < 16) ? 0 : 16));
}

// conv variant: per-lane source column may fall outside the image -> zeros
__device__ __forceinline__ v16h load_bT_conv(const _Float16* Xrow, int Cin,
                                             int k0, int sxbase) {
  int lane = threadIdx.x & 31;
  int sx = sxbase + (lane & 15);
  int kb = k0 + ((lane < 16) ? 0 : 16);
  v16h b = {};
  if (sx >= 0 && sx < NW) b = *(const v16h*)(Xrow + (size_t)sx * Cin + kb);
  return b;
}

// pack 8 f32 accumulator rows into 8 contiguous f16 (one 16B store)
__device__ __forceinline__ v8h pack8(v8f a) {
  v8h r;
#pragma unroll
  for (int i = 0; i < 8; ++i) r[i] = (_Float16)a[i];
  return r;
}

// cross-lane reductions within each 16-lane half (rows of C layout)
__device__ __forceinline__ float hmax16(float v) {
#pragma unroll
  for (int m = 1; m < 16; m <<= 1) v = fmaxf(v, __shfl_xor(v, m, 32));
  return v;
}
__device__ __forceinline__ float hsum16(float v) {
#pragma unroll
  for (int m = 1; m < 16; m <<= 1) v += __shfl_xor(v, m, 32);
  return v;
}

// ---------------------------------------------------------------------------
// Packing / conversion kernels
// ---------------------------------------------------------------------------
__global__ void k_f32_to_f16(const float* __restrict__ in,
                             _Float16* __restrict__ out, int n) {
  int i = blockIdx.x * blockDim.x + threadIdx.x;
  if (i < n) out[i] = (_Float16)in[i];
}

// (B,C,S) f32 -> pixel-major (B,S,C) f16
__global__ void k_xpose(const float* __restrict__ in, _Float16* __restrict__ out) {
  int idx = blockIdx.x * blockDim.x + threadIdx.x;
  if (idx >= NB * NC * NS) return;
  int c = idx % NC;
  int s = (idx / NC) % NS;
  int b = idx / (NC * NS);
  out[idx] = (_Float16)in[((size_t)b * NC + c) * NS + s];
}

// conv weight (Co,Ci,3,3) -> per-tap row-major A f16 [9][Co][Ci]
__global__ void k_pack_convw(const float* __restrict__ w,
                             _Float16* __restrict__ out, int Co, int Ci) {
  int idx = blockIdx.x * blockDim.x + threadIdx.x;
  int total = 9 * Co * Ci;
  if (idx >= total) return;
  int ci  = idx % Ci;
  int o   = (idx / Ci) % Co;
  int tap = idx / (Ci * Co);
  int ky = tap / 3, kx = tap % 3;
  out[idx] = (_Float16)w[(((size_t)o * Ci + ci) * 3 + ky) * 3 + kx];
}

// BN folding: scale = g*rsqrt(v+eps), shift = b - m*scale
__global__ void k_affine(const float* __restrict__ g, const float* __restrict__ b,
                         const float* __restrict__ m, const float* __restrict__ v,
                         float* __restrict__ scale, float* __restrict__ shift,
                         int n) {
  int i = blockIdx.x * blockDim.x + threadIdx.x;
  if (i >= n) return;
  float s = g[i] * rsqrtf(v[i] + EPSV);
  scale[i] = s;
  shift[i] = b[i] - m[i] * s;
}

// ---------------------------------------------------------------------------
// QKV projection: per (branch, which, batch) GEMM  M=512, N=1024, K=512
// each wave computes a 32x32 output tile (4 accumulators, frags reused 2x)
// Q -> (S,D) * 1/sqrt(D) ; K -> (S,D) ; V -> (D,S)
// ---------------------------------------------------------------------------
__global__ void k_qkv(const _Float16* __restrict__ xh,   // [3][B][S][C] pixel-major
                      const _Float16* __restrict__ wq,   // [3][512][512]
                      const _Float16* __restrict__ wk,
                      const _Float16* __restrict__ wv,
                      const float* __restrict__ bq, const float* __restrict__ bk,
                      const float* __restrict__ bv,
                      _Float16* __restrict__ qb,  // [3][B][16][1024][32]
                      _Float16* __restrict__ kb,  // [3][B][16][1024][32]
                      _Float16* __restrict__ vb)  // [3][B][512][1024]
{
  int id = blockIdx.x;
  int nt = id & 31;
  int mt = (id >> 5) & 15;
  int rest = id >> 9;
  int b = rest & 1; rest >>= 1;
  int which = rest % 3;
  int br    = rest / 3;
  int lane = threadIdx.x & 31;

  int inp = (which == 0) ? (2 - br) : 1;  // q from "other", k/v from x
  const _Float16* Xt = xh + ((size_t)inp * NB + b) * NS * NC;  // [S][C]
  const _Float16* W;
  const float* bias;
  if (which == 0)      { W = wq + (size_t)br * NC * NC; bias = bq + br * NC; }
  else if (which == 1) { W = wk + (size_t)br * NC * NC; bias = bk + br * NC; }
  else                 { W = wv + (size_t)br * NC * NC; bias = bv + br * NC; }

  int m0 = mt * 32, n0 = nt * 32;
  v8f acc[2][2] = {};
#pragma unroll 4
  for (int kk = 0; kk < NC / 32; ++kk) {
    v16h a0 = load_a_frag(W + (size_t)m0 * NC + kk * 32, NC);
    v16h a1 = load_a_frag(W + (size_t)(m0 + 16) * NC + kk * 32, NC);
    v16h b0 = load_bT_frag(Xt + (size_t)n0 * NC + kk * 32, NC);
    v16h b1 = load_bT_frag(Xt + (size_t)(n0 + 16) * NC + kk * 32, NC);
    acc[0][0] = wmma32(a0, b0, acc[0][0]);
    acc[0][1] = wmma32(a0, b1, acc[0][1]);
    acc[1][0] = wmma32(a1, b0, acc[1][0]);
    acc[1][1] = wmma32(a1, b1, acc[1][1]);
  }

  int col = lane & 15;
  int mo  = (lane < 16) ? 0 : 8;
  size_t bb = (size_t)br * NB + b;
  float qs = (which == 0) ? 0.17677669529663687f : 1.0f;  // 1/sqrt(32)
#pragma unroll
  for (int ms = 0; ms < 2; ++ms)
#pragma unroll
    for (int ns = 0; ns < 2; ++ns) {
      int ncol  = n0 + ns * 16 + col;
      int mbase = m0 + ms * 16 + mo;       // multiple of 8
      if (which == 2) {
        // V: (D,S) channel-major -> strided scalar stores
#pragma unroll
        for (int r = 0; r < 8; ++r) {
          int mrow = mbase + r;
          vb[(bb * NC + mrow) * NS + ncol] = (_Float16)(acc[ms][ns][r] + bias[mrow]);
        }
      } else {
        // Q/K: (S,D) -> 8 consecutive d per lane -> one 16B store
        v8f t = acc[ms][ns];
#pragma unroll
        for (int r = 0; r < 8; ++r) t[r] = (t[r] + bias[mbase + r]) * qs;
        int h = mbase >> 5, d0 = mbase & 31;
        _Float16* dst = which ? kb : qb;
        *(v8h*)(dst + ((bb * NHD + h) * NS + ncol) * ND + d0) = pack8(t);
      }
    }
}

// ---------------------------------------------------------------------------
// Flash attention + per-head output projection + branch BN
// one wave per (branch, batch, head, 16-row tile of S); all frags contiguous
// ---------------------------------------------------------------------------
__global__ void k_attn(const _Float16* __restrict__ qb,    // [..][S][D]
                       const _Float16* __restrict__ kbuf,  // [..][S][D]
                       const _Float16* __restrict__ vbuf,  // [..][D][S]
                       const _Float16* __restrict__ woh,   // [3][16][32][32] (i,j)
                       const float* __restrict__ bo,       // [3][512]
                       const float* __restrict__ bsc, const float* __restrict__ bsh,
                       _Float16* __restrict__ xt)          // [B][S][1536] pixel-major
{
  __shared__ _Float16 pt[16 * 32];  // prob tile bounce (C-layout -> A-layout)
  __shared__ _Float16 yt[16 * 32];  // y tile bounce

  int id = blockIdx.x;
  int mt = id & 63; id >>= 6;
  int h  = id & 15; id >>= 4;
  int b  = id & 1;  id >>= 1;
  int br = id;
  int lane = threadIdx.x & 31;
  size_t bb = (size_t)br * NB + b;

  const _Float16* Q  = qb   + ((bb * NHD + h) * NS) * ND;
  const _Float16* K  = kbuf + ((bb * NHD + h) * NS) * ND;
  const _Float16* Vt = vbuf + (bb * NC + h * ND) * NS;  // (D,S)
  int m0 = mt * 16;

  v16h aq = load_a_frag(Q + (size_t)m0 * ND, ND);

  float mx[8], sm[8];
#pragma unroll
  for (int r = 0; r < 8; ++r) { mx[r] = -1e30f; sm[r] = 0.0f; }
  v8f acc0 = {}, acc1 = {};
  v8f z = {};

  int col = lane & 15;
  int mo  = (lane < 16) ? 0 : 8;

  for (int it = 0; it < NS / 32; ++it) {
    int t0 = it * 32;
    // scores: B-frag = contiguous rows of K (S,D)
    v16h bke = load_bT_frag(K + (size_t)t0 * ND, ND);
    v16h bko = load_bT_frag(K + (size_t)(t0 + 16) * ND, ND);
    v8f ce = wmma32(aq, bke, z);
    v8f co = wmma32(aq, bko, z);
#pragma unroll
    for (int r = 0; r < 8; ++r) {
      float tmx = hmax16(fmaxf(ce[r], co[r]));
      float nm  = fmaxf(mx[r], tmx);
      float rs  = __expf(mx[r] - nm);
      float pe  = __expf(ce[r] - nm);
      float po  = __expf(co[r] - nm);
      sm[r] = sm[r] * rs + hsum16(pe + po);
      acc0[r] *= rs; acc1[r] *= rs;
      mx[r] = nm;
      pt[(mo + r) * 32 + col]      = (_Float16)pe;
      pt[(mo + r) * 32 + 16 + col] = (_Float16)po;
    }
    v16h ap = load_a_frag(pt, 32);
    // AV: B-frag = contiguous rows of V (D,S)
    v16h b0 = load_bT_frag(Vt + t0, NS);
    v16h b1 = load_bT_frag(Vt + (size_t)16 * NS + t0, NS);
    acc0 = wmma32(ap, b0, acc0);
    acc1 = wmma32(ap, b1, acc1);
  }

  // normalize rows, bounce y through LDS to re-shape as A fragment
#pragma unroll
  for (int r = 0; r < 8; ++r) {
    float iv = 1.0f / sm[r];
    yt[(mo + r) * 32 + col]      = (_Float16)(acc0[r] * iv);
    yt[(mo + r) * 32 + 16 + col] = (_Float16)(acc1[r] * iv);
  }

  // per-head output projection: o[s,i] = sum_j y[s,j] * Wo[h][i][j]
  // B-frag = contiguous rows of Wo (i-major, j contiguous)
  const _Float16* Wt = woh + ((size_t)br * NHD + h) * ND * ND;
  v16h ay = load_a_frag(yt, 32);
  v16h w0 = load_bT_frag(Wt, ND);
  v16h w1 = load_bT_frag(Wt + (size_t)16 * ND, ND);
  v8f o0 = wmma32(ay, w0, z);
  v8f o1 = wmma32(ay, w1, z);

  // bias + BN epilogue -> concat buffer (B, S, 3C) pixel-major f16
#pragma unroll
  for (int half = 0; half < 2; ++half) {
    v8f oo = half ? o1 : o0;
    int ci = h * ND + half * 16 + col;
    int gc = br * NC + ci;
    float s_ = bsc[gc], t_ = bsh[gc], b_ = bo[gc];
#pragma unroll
    for (int r = 0; r < 8; ++r) {
      int srow = m0 + mo + r;
      float val = (oo[r] + b_) * s_ + t_;
      xt[((size_t)b * NS + srow) * (3 * NC) + gc] = (_Float16)val;
    }
  }
}

// ---------------------------------------------------------------------------
// 3x3 SAME conv as implicit GEMM (9 shifted K=Cin GEMMs) + BN + ReLU
// one wave per (batch, 32 out-channels, image row); 32x32 tile per wave
// ---------------------------------------------------------------------------
__global__ void k_conv3(const _Float16* __restrict__ Ataps,  // [9][512][Cin]
                        const _Float16* __restrict__ Xall,   // [B][S][Cin] pix-major
                        int Cin,
                        const float* __restrict__ scale,
                        const float* __restrict__ shift,
                        _Float16* __restrict__ outh,  // f16 [B][S][512] pix-major
                        float* __restrict__ outf,     // f32 (B,512,32,32) final
                        int final_out) {
  int id = blockIdx.x;
  int y  = id & 31;
  int mt = (id >> 5) & 15;
  int b  = id >> 9;
  int lane = threadIdx.x & 31;

  int m0 = mt * 32;
  v8f acc[2][2] = {};

  for (int ky = 0; ky < 3; ++ky) {
    int yy = y + ky - 1;
    if ((unsigned)yy >= (unsigned)NW) continue;
    const _Float16* Xrow = Xall + ((size_t)b * NS + yy * NW) * Cin;
    for (int kx = 0; kx < 3; ++kx) {
      int tap = ky * 3 + kx;
      int dx  = kx - 1;
      const _Float16* A = Ataps + ((size_t)tap * NC + m0) * Cin;
      for (int kk = 0; kk < Cin / 32; ++kk) {
        __builtin_prefetch(A + (size_t)(kk + 1) * 32, 0, 1);
        v16h a0 = load_a_frag(A + kk * 32, Cin);
        v16h a1 = load_a_frag(A + (size_t)16 * Cin + kk * 32, Cin);
        v16h b0 = load_bT_conv(Xrow, Cin, kk * 32, dx);
        v16h b1 = load_bT_conv(Xrow, Cin, kk * 32, 16 + dx);
        acc[0][0] = wmma32(a0, b0, acc[0][0]);
        acc[0][1] = wmma32(a0, b1, acc[0][1]);
        acc[1][0] = wmma32(a1, b0, acc[1][0]);
        acc[1][1] = wmma32(a1, b1, acc[1][1]);
      }
    }
  }

  int col = lane & 15;
  int mo  = (lane < 16) ? 0 : 8;
#pragma unroll
  for (int ms = 0; ms < 2; ++ms)
#pragma unroll
    for (int ns = 0; ns < 2; ++ns) {
      int n = y * NW + ns * 16 + col;
      int o0 = m0 + ms * 16 + mo;  // multiple of 8
      v8f t = acc[ms][ns];
#pragma unroll
      for (int r = 0; r < 8; ++r)
        t[r] = fmaxf(t[r] * scale[o0 + r] + shift[o0 + r], 0.0f);
      if (final_out) {
        // final output is channel-major f32 -> strided scalar stores
#pragma unroll
        for (int r = 0; r < 8; ++r)
          outf[((size_t)b * NC + o0 + r) * NS + n] = t[r];
      } else {
        // pixel-major f16 intermediate: 8 consecutive channels -> one 16B store
        *(v8h*)(outh + ((size_t)b * NS + n) * NC + o0) = pack8(t);
      }
    }
}

// ---------------------------------------------------------------------------
// Host launcher
// ---------------------------------------------------------------------------
extern "C" void kernel_launch(void* const* d_in, const int* in_sizes, int n_in,
                              void* d_out, int out_size, void* d_ws, size_t ws_size,
                              hipStream_t stream) {
  const float* x_prev = (const float*)d_in[0];
  const float* x      = (const float*)d_in[1];
  const float* x_next = (const float*)d_in[2];
  const float* Wq = (const float*)d_in[3];  const float* bq = (const float*)d_in[4];
  const float* Wk = (const float*)d_in[5];  const float* bk = (const float*)d_in[6];
  const float* Wv = (const float*)d_in[7];  const float* bv = (const float*)d_in[8];
  const float* Wo = (const float*)d_in[9];  const float* bo = (const float*)d_in[10];
  const float* bng = (const float*)d_in[11]; const float* bnb = (const float*)d_in[12];
  const float* bnm = (const float*)d_in[13]; const float* bnv = (const float*)d_in[14];
  const float* c1w = (const float*)d_in[15];
  const float* bn1g = (const float*)d_in[16]; const float* bn1b = (const float*)d_in[17];
  const float* bn1m = (const float*)d_in[18]; const float* bn1v = (const float*)d_in[19];
  const float* c2w = (const float*)d_in[20];
  const float* bn2g = (const float*)d_in[21]; const float* bn2b = (const float*)d_in[22];
  const float* bn2m = (const float*)d_in[23]; const float* bn2v = (const float*)d_in[24];

  char* wp = (char*)d_ws;
  auto alloc = [&](size_t bytes) -> void* {
    void* p = (void*)wp;
    wp += (bytes + 255) & ~(size_t)255;
    return p;
  };
  const size_t BCS = (size_t)NB * NC * NS;  // 1,048,576
  const size_t WSZ = (size_t)3 * NC * NC;   // 786,432

  _Float16* xh   = (_Float16*)alloc(3 * BCS * 2);   // pixel-major activations
  _Float16* wqh  = (_Float16*)alloc(WSZ * 2);
  _Float16* wkh  = (_Float16*)alloc(WSZ * 2);
  _Float16* wvh  = (_Float16*)alloc(WSZ * 2);
  _Float16* woh  = (_Float16*)alloc((size_t)3 * NHD * ND * ND * 2);
  _Float16* a1   = (_Float16*)alloc((size_t)9 * NC * (3 * NC) * 2);
  _Float16* a2   = (_Float16*)alloc((size_t)9 * NC * NC * 2);
  _Float16* qbuf = (_Float16*)alloc(3 * BCS * 2);
  _Float16* kbuf = (_Float16*)alloc(3 * BCS * 2);
  _Float16* vbuf = (_Float16*)alloc(3 * BCS * 2);
  _Float16* xt   = (_Float16*)alloc((size_t)NB * 3 * NC * NS * 2);
  _Float16* h1   = (_Float16*)alloc(BCS * 2);
  float* bsc = (float*)alloc(3 * NC * 4);
  float* bsh = (float*)alloc(3 * NC * 4);
  float* c1s = (float*)alloc(NC * 4);
  float* c1t = (float*)alloc(NC * 4);
  float* c2s = (float*)alloc(NC * 4);
  float* c2t = (float*)alloc(NC * 4);

  auto g1 = [](size_t n) { return (unsigned)((n + 255) / 256); };

  // pack / convert
  k_xpose<<<g1(BCS), 256, 0, stream>>>(x_prev, xh + 0 * BCS);
  k_xpose<<<g1(BCS), 256, 0, stream>>>(x,      xh + 1 * BCS);
  k_xpose<<<g1(BCS), 256, 0, stream>>>(x_next, xh + 2 * BCS);
  k_f32_to_f16<<<g1(WSZ), 256, 0, stream>>>(Wq, wqh, (int)WSZ);
  k_f32_to_f16<<<g1(WSZ), 256, 0, stream>>>(Wk, wkh, (int)WSZ);
  k_f32_to_f16<<<g1(WSZ), 256, 0, stream>>>(Wv, wvh, (int)WSZ);
  k_f32_to_f16<<<g1(3 * NHD * ND * ND), 256, 0, stream>>>(Wo, woh,
                                                          3 * NHD * ND * ND);
  k_pack_convw<<<g1((size_t)9 * NC * 3 * NC), 256, 0, stream>>>(c1w, a1, NC, 3 * NC);
  k_pack_convw<<<g1((size_t)9 * NC * NC), 256, 0, stream>>>(c2w, a2, NC, NC);
  k_affine<<<g1(3 * NC), 256, 0, stream>>>(bng, bnb, bnm, bnv, bsc, bsh, 3 * NC);
  k_affine<<<g1(NC), 256, 0, stream>>>(bn1g, bn1b, bn1m, bn1v, c1s, c1t, NC);
  k_affine<<<g1(NC), 256, 0, stream>>>(bn2g, bn2b, bn2m, bn2v, c2s, c2t, NC);

  // QKV: 3 branches x {q,k,v} x B x 16 m-tiles(32) x 32 n-tiles(32)
  k_qkv<<<3 * 3 * NB * 16 * 32, 32, 0, stream>>>(xh, wqh, wkh, wvh, bq, bk, bv,
                                                 qbuf, kbuf, vbuf);
  // attention + out-proj + branch BN
  k_attn<<<3 * NB * NHD * 64, 32, 0, stream>>>(qbuf, kbuf, vbuf, woh, bo, bsc,
                                               bsh, xt);
  // DoubleConv
  k_conv3<<<NB * 16 * 32, 32, 0, stream>>>(a1, xt, 3 * NC, c1s, c1t, h1, nullptr, 0);
  k_conv3<<<NB * 16 * 32, 32, 0, stream>>>(a2, h1, NC, c2s, c2t, nullptr,
                                           (float*)d_out, 1);
}
